// LSTMDecoder_85306640433774
// MI455X (gfx1250) — matrix operand order repaired and checked by reading, not compiled, not run on hardware
//
#include <hip/hip_runtime.h>

// ---------------------------------------------------------------------------
// LSTM decoder for MI455X (gfx1250): bf16 WMMA (v_wmma_f32_16x16x32_bf16),
// wave32. Each wave owns a 32x16 output macro-tile (2 M-tiles) so every
// B (weight) fragment feeds 2 WMMAs -> halves L2 weight traffic.
// Time-invariant x-path is hoisted out of the time loop.
// ---------------------------------------------------------------------------

typedef __attribute__((ext_vector_type(16))) __bf16 v16bf;
typedef __attribute__((ext_vector_type(8)))  __bf16 v8bf;
typedef __attribute__((ext_vector_type(8)))  float  v8f;

#define B_   1024
#define IN_  256
#define H_   512
#define OUT_ 256
#define SEQ_ 64
#define G4H  (4 * H_)

// ---- helpers --------------------------------------------------------------

__device__ __forceinline__ __bf16 f2bf(float f) {
  unsigned u = __builtin_bit_cast(unsigned, f);
  u += 0x7FFFu + ((u >> 16) & 1u);              // round-to-nearest-even
  unsigned short s = (unsigned short)(u >> 16);
  return __builtin_bit_cast(__bf16, s);
}

__device__ __forceinline__ float fast_sigmoid(float x) {
  return 1.0f / (1.0f + __expf(-x));
}

__device__ __forceinline__ float fast_tanh(float x) {
  x = fminf(fmaxf(x, -15.0f), 15.0f);           // avoid inf/inf
  float e = __expf(-2.0f * x);
  return (1.0f - e) / (1.0f + e);
}

__device__ __forceinline__ v8f wmma_bf16(v16bf a, v16bf b, v8f c) {
  // D = A(16x32 bf16) * B(32x16 bf16) + C(16x16 f32)
  return __builtin_amdgcn_wmma_f32_16x16x32_bf16(
      /*neg_a=*/false, a, /*neg_b=*/false, b,
      /*c_mod=*/(short)0, c, /*reuse_a=*/false, /*reuse_b=*/false);
}

// A fragment: 16x32 bf16 tile of row-major A at (m0, k0), leading dim lda.
// Lane l (l<16): row m0+l, K {k0..k0+15}; lanes 16-31 shift K by 16.
__device__ __forceinline__ v16bf load_a_frag(const __bf16* __restrict__ base,
                                             int lda, int m0, int k0) {
  const int lane = threadIdx.x & 31;
  const int row  = m0 + (lane & 15);
  const int kk   = k0 + ((lane >> 4) << 4);
  const __bf16* p = base + (size_t)row * lda + kk;
  v8bf lo = *(const v8bf*)(p);
  v8bf hi = *(const v8bf*)(p + 8);
  return __builtin_shufflevector(lo, hi, 0,1,2,3,4,5,6,7,8,9,10,11,12,13,14,15);
}

// B fragment: 32x16 tile of B = W^T, where W is (N x K) row-major.
// B[k][n] = W[n*ldw + k]; lane l holds col n0+(l&15), 16 contiguous K values.
__device__ __forceinline__ v16bf load_b_frag(const __bf16* __restrict__ wt,
                                             int ldw, int n0, int k0) {
  const int lane = threadIdx.x & 31;
  const int col  = n0 + (lane & 15);
  const int kk   = k0 + ((lane >> 4) << 4);
  return *(const v16bf*)(wt + (size_t)col * ldw + kk);
}

// ---- small utility kernels ------------------------------------------------

__global__ void cvt_bf16_kernel(const float* __restrict__ src,
                                __bf16* __restrict__ dst, int n) {
  int i = blockIdx.x * blockDim.x + threadIdx.x;
  if (i < n) dst[i] = f2bf(src[i]);
}

__global__ void bias_sum_kernel(const float* __restrict__ a,
                                const float* __restrict__ b,
                                float* __restrict__ o, int n) {
  int i = blockIdx.x * blockDim.x + threadIdx.x;
  if (i < n) o[i] = a[i] + b[i];
}

__global__ void init_state_kernel(const float* __restrict__ h0,
                                  const float* __restrict__ c0,
                                  float* __restrict__ hs,
                                  float* __restrict__ cs,
                                  __bf16* __restrict__ hbf, int n) {
  int i = blockIdx.x * blockDim.x + threadIdx.x;
  if (i < n) {
    float h = h0[i];
    hs[i]  = h;
    hbf[i] = f2bf(h);
    cs[i]  = c0[i];
  }
}

// ---- GEMM: out = A(1024 x K) * W^T + bias, one wave per 32x16 macro-tile --

__global__ __launch_bounds__(256)
void gemm_bf16_bias_kernel(const __bf16* __restrict__ A,
                           const __bf16* __restrict__ WT,   // (N x K) row-major
                           const float* __restrict__ bias,  // (N)
                           float* __restrict__ out,
                           int K, int n_shift, int out_row_stride) {
  const int wave = (int)((blockIdx.x * blockDim.x + threadIdx.x) >> 5);
  const int m0 = (wave >> n_shift) << 5;                    // 32 rows / wave
  const int n0 = (wave & ((1 << n_shift) - 1)) << 4;
  const int lane = threadIdx.x & 31;
  const int colL = lane & 15;
  const int rowBase = (lane >> 4) << 3;

  float bv = bias[n0 + colL];
  v8f acc[2];
#pragma unroll
  for (int mt = 0; mt < 2; ++mt)
#pragma unroll
    for (int r = 0; r < 8; ++r) acc[mt][r] = bv;

#pragma unroll 4
  for (int kc = 0; kc < K; kc += 32) {
    v16bf b  = load_b_frag(WT, K, n0, kc);
    v16bf a0 = load_a_frag(A, K, m0, kc);
    v16bf a1 = load_a_frag(A, K, m0 + 16, kc);
    acc[0] = wmma_bf16(a0, b, acc[0]);
    acc[1] = wmma_bf16(a1, b, acc[1]);
  }

#pragma unroll
  for (int mt = 0; mt < 2; ++mt) {
    float* o = out + (size_t)(m0 + mt * 16 + rowBase) * out_row_stride + n0 + colL;
#pragma unroll
    for (int r = 0; r < 8; ++r) o[(size_t)r * out_row_stride] = acc[mt][r];
  }
}

// ---- fused LSTM cell: 4 gate GEMM macro-tiles + pointwise update ----------
// One wave owns a 32x16 h macro-tile (m0,n0): accumulates the 4 gate tiles
// at columns n0 + g*H (g = i,f,g,o) for both M-tiles, then updates c/h and
// emits bf16 h for the next step's GEMMs.

template <bool HAS_X>
__global__ __launch_bounds__(256)
void lstm_cell_kernel(const __bf16* __restrict__ Axbf,  // x input (bf16) [HAS_X]
                      const __bf16* __restrict__ Wxbf,  // (4H x H)        [HAS_X]
                      const __bf16* __restrict__ Ahbf,  // h prev (bf16)
                      const __bf16* __restrict__ Whbf,  // (4H x H)
                      const float*  __restrict__ pre,   // (B x 4H)       [!HAS_X]
                      const float*  __restrict__ bsum,  // (4H)            [HAS_X]
                      float* __restrict__ cstate,
                      float* __restrict__ hstate,
                      __bf16* __restrict__ hbf_out) {
  const int wave = (int)((blockIdx.x * blockDim.x + threadIdx.x) >> 5);
  const int m0 = (wave >> 5) << 5;   // 32 row-pair tiles (B = 1024)
  const int n0 = (wave & 31) << 4;   // 32 col tiles (H = 512)
  const int lane = threadIdx.x & 31;
  const int colL = lane & 15;
  const int rowBase = (lane >> 4) << 3;

  v8f acc[2][4];
  if constexpr (HAS_X) {
#pragma unroll
    for (int g = 0; g < 4; ++g) {
      float bv = bsum[g * H_ + n0 + colL];
#pragma unroll
      for (int mt = 0; mt < 2; ++mt)
#pragma unroll
        for (int r = 0; r < 8; ++r) acc[mt][g][r] = bv;
    }
  } else {
#pragma unroll
    for (int mt = 0; mt < 2; ++mt)
#pragma unroll
      for (int g = 0; g < 4; ++g) {
        const float* p =
            pre + (size_t)(m0 + mt * 16 + rowBase) * G4H + g * H_ + n0 + colL;
#pragma unroll
        for (int r = 0; r < 8; ++r) acc[mt][g][r] = p[(size_t)r * G4H];
      }
  }

#pragma unroll 2
  for (int kc = 0; kc < H_; kc += 32) {
    v16bf ah0 = load_a_frag(Ahbf, H_, m0, kc);
    v16bf ah1 = load_a_frag(Ahbf, H_, m0 + 16, kc);
    v16bf ax0, ax1;
    if constexpr (HAS_X) {
      ax0 = load_a_frag(Axbf, H_, m0, kc);
      ax1 = load_a_frag(Axbf, H_, m0 + 16, kc);
    }
#pragma unroll
    for (int g = 0; g < 4; ++g) {
      if constexpr (HAS_X) {
        v16bf bx = load_b_frag(Wxbf, H_, g * H_ + n0, kc);
        acc[0][g] = wmma_bf16(ax0, bx, acc[0][g]);
        acc[1][g] = wmma_bf16(ax1, bx, acc[1][g]);
      }
      v16bf bh = load_b_frag(Whbf, H_, g * H_ + n0, kc);
      acc[0][g] = wmma_bf16(ah0, bh, acc[0][g]);
      acc[1][g] = wmma_bf16(ah1, bh, acc[1][g]);
    }
  }

#pragma unroll
  for (int mt = 0; mt < 2; ++mt)
#pragma unroll
    for (int r = 0; r < 8; ++r) {
      const int idx = (m0 + mt * 16 + rowBase + r) * H_ + n0 + colL;
      float ig = fast_sigmoid(acc[mt][0][r]);
      float fg = fast_sigmoid(acc[mt][1][r]);
      float gg = fast_tanh(acc[mt][2][r]);
      float og = fast_sigmoid(acc[mt][3][r]);
      float cn = fg * cstate[idx] + ig * gg;
      float hn = og * fast_tanh(cn);
      cstate[idx]  = cn;
      hstate[idx]  = hn;
      hbf_out[idx] = f2bf(hn);
    }
}

// ---- host-side launch -----------------------------------------------------

extern "C" void kernel_launch(void* const* d_in, const int* in_sizes, int n_in,
                              void* d_out, int out_size, void* d_ws, size_t ws_size,
                              hipStream_t stream) {
  const float* x    = (const float*)d_in[0];
  const float* h0   = (const float*)d_in[1];
  const float* c0   = (const float*)d_in[2];
  const float* Wih0 = (const float*)d_in[3];
  const float* Whh0 = (const float*)d_in[4];
  const float* bih0 = (const float*)d_in[5];
  const float* bhh0 = (const float*)d_in[6];
  const float* Wih1 = (const float*)d_in[7];
  const float* Whh1 = (const float*)d_in[8];
  const float* bih1 = (const float*)d_in[9];
  const float* bhh1 = (const float*)d_in[10];
  const float* Wfc  = (const float*)d_in[11];
  const float* bfc  = (const float*)d_in[12];

  float* outp    = (float*)d_out;
  float* decoded = outp;                               // (B, SEQ, OUT)
  float* hT      = outp + (size_t)B_ * SEQ_ * OUT_;    // (L, B, H) live h-state
  float* cT      = hT + (size_t)2 * B_ * H_;           // (L, B, H) live c-state
  float* h0s = hT;
  float* h1s = hT + (size_t)B_ * H_;
  float* c0s = cT;
  float* c1s = cT + (size_t)B_ * H_;

  char* ws = (char*)d_ws;
  size_t off = 0;
  auto alloc = [&](size_t bytes) -> char* {
    char* p = ws + off;
    off += (bytes + 255) & ~(size_t)255;
    return p;
  };
  __bf16* xbf    = (__bf16*)alloc((size_t)B_ * IN_ * 2);
  __bf16* Wih0bf = (__bf16*)alloc((size_t)G4H * IN_ * 2);
  __bf16* Whh0bf = (__bf16*)alloc((size_t)G4H * H_ * 2);
  __bf16* Wih1bf = (__bf16*)alloc((size_t)G4H * H_ * 2);
  __bf16* Whh1bf = (__bf16*)alloc((size_t)G4H * H_ * 2);
  __bf16* Wfcbf  = (__bf16*)alloc((size_t)OUT_ * H_ * 2);
  float*  bsum0  = (float*)alloc((size_t)G4H * 4);
  float*  bsum1  = (float*)alloc((size_t)G4H * 4);
  float*  pre0   = (float*)alloc((size_t)B_ * G4H * 4);
  __bf16* hbfA   = (__bf16*)alloc((size_t)2 * B_ * H_ * 2);  // double-buffered
  __bf16* hbfB   = (__bf16*)alloc((size_t)2 * B_ * H_ * 2);

  auto cvt = [&](const float* s, __bf16* d, int n) {
    cvt_bf16_kernel<<<(n + 255) / 256, 256, 0, stream>>>(s, d, n);
  };
  cvt(x,    xbf,    B_ * IN_);
  cvt(Wih0, Wih0bf, G4H * IN_);
  cvt(Whh0, Whh0bf, G4H * H_);
  cvt(Wih1, Wih1bf, G4H * H_);
  cvt(Whh1, Whh1bf, G4H * H_);
  cvt(Wfc,  Wfcbf,  OUT_ * H_);

  bias_sum_kernel<<<(G4H + 255) / 256, 256, 0, stream>>>(bih0, bhh0, bsum0, G4H);
  bias_sum_kernel<<<(G4H + 255) / 256, 256, 0, stream>>>(bih1, bhh1, bsum1, G4H);

  {
    int n = 2 * B_ * H_;
    init_state_kernel<<<(n + 255) / 256, 256, 0, stream>>>(h0, c0, hT, cT, hbfA, n);
  }

  // pre0 = x @ Wih0^T + (bih0 + bhh0): M=1024, N=2048, K=256
  // macro-tiles: (1024/32) x (2048/16) = 32 x 128 -> 4096 waves
  gemm_bf16_bias_kernel<<<4096 / 8, 256, 0, stream>>>(
      xbf, Wih0bf, bsum0, pre0, IN_, /*n_shift=*/7, /*ldc=*/G4H);

  __bf16* cur = hbfA;   // bf16 h state at step t-1
  __bf16* nxt = hbfB;   // bf16 h state at step t
  for (int t = 0; t < SEQ_; ++t) {
    __bf16* h0_in  = cur;
    __bf16* h1_in  = cur + (size_t)B_ * H_;
    __bf16* h0_out = nxt;
    __bf16* h1_out = nxt + (size_t)B_ * H_;

    // layer 0: gates = pre0 + h0 @ Whh0^T  (32 x 32 macro-tiles = 1024 waves)
    lstm_cell_kernel<false><<<1024 / 8, 256, 0, stream>>>(
        nullptr, nullptr, h0_in, Whh0bf, pre0, nullptr, c0s, h0s, h0_out);

    // layer 1: gates = bsum1 + h0(t) @ Wih1^T + h1 @ Whh1^T  (1024 waves)
    lstm_cell_kernel<true><<<1024 / 8, 256, 0, stream>>>(
        h0_out, Wih1bf, h1_in, Whh1bf, nullptr, bsum1, c1s, h1s, h1_out);

    // FC head: decoded[:, t, :] = h1(t) @ Wfc^T + bfc
    // macro-tiles: (1024/32) x (256/16) = 32 x 16 -> 512 waves
    gemm_bf16_bias_kernel<<<512 / 8, 256, 0, stream>>>(
        h1_out, Wfcbf, bfc, decoded + (size_t)t * OUT_,
        H_, /*n_shift=*/4, /*ldc=*/SEQ_ * OUT_);

    __bf16* tmp = cur; cur = nxt; nxt = tmp;
  }
}
